// CapsuleNet_1546188227172
// MI455X (gfx1250) — matrix operand order repaired and checked by reading, not compiled, not run on hardware
//
#include <hip/hip_runtime.h>

// ---------------- problem constants ----------------
#define NN      30000           // nodes
#define MM      16              // neighbors
#define NFEAT   1024
#define NCLASS  16
#define KC      8               // capsules
#define DD      32              // per-capsule dim  (== wave32!)
#define DDIM    256             // KC*DD
#define NLAYER  3

typedef __attribute__((ext_vector_type(16))) __bf16    v16bf;
typedef __attribute__((ext_vector_type(8)))  __bf16    v8bf;
typedef __attribute__((ext_vector_type(8)))  float     v8f;
typedef __attribute__((ext_vector_type(4)))  float     v4f;
typedef __attribute__((ext_vector_type(4)))  unsigned  v4u;
typedef __attribute__((ext_vector_type(4)))  int       v4i;
typedef __attribute__((ext_vector_type(8)))  int       v8i;

#if __has_builtin(__builtin_amdgcn_tensor_load_to_lds) && \
    __has_builtin(__builtin_amdgcn_s_wait_tensorcnt)
#define HAVE_TDM 1
#else
#define HAVE_TDM 0
#endif

// exact 32-lane butterfly reduction (wave32)
__device__ __forceinline__ float wave_sum(float v) {
    v += __shfl_xor(v, 16, 32);
    v += __shfl_xor(v, 8, 32);
    v += __shfl_xor(v, 4, 32);
    v += __shfl_xor(v, 2, 32);
    v += __shfl_xor(v, 1, 32);
    return v;
}

// ---------------------------------------------------------------------------
// Kernel 0: transpose + convert pca_w [1024x256] f32 -> Wt [256x1024] bf16
// ---------------------------------------------------------------------------
__global__ void k_conv_wt(const float* __restrict__ W, __bf16* __restrict__ Wt) {
    int id = blockIdx.x * 256 + threadIdx.x;       // 0 .. 256*1024-1
    int n = id >> 10;
    int k = id & 1023;
    Wt[id] = (__bf16)W[(size_t)k * DDIM + n];
}

// ---------------------------------------------------------------------------
// Kernel 1: H = relu(X @ W + b)   via v_wmma_f32_16x16x32_bf16
//   One block (8 waves) per 16-row M-panel. A panel (16x1024) converted to
//   bf16 in LDS once, reused by all waves. Each wave owns TWO 16x16 N-tiles.
// ---------------------------------------------------------------------------
__global__ void k_pca_gemm(const float* __restrict__ X,
                           const __bf16* __restrict__ Wt,
                           const float* __restrict__ bias,
                           float* __restrict__ H) {
    __shared__ __bf16 As[16 * NFEAT];    // 32 KB A panel (bf16)
    const int t = threadIdx.x;
    const int mtile = blockIdx.x;        // 0..1874

    // cooperative fp32 -> bf16 convert of A panel (coalesced float4 loads)
    for (int i = t * 4; i < 16 * NFEAT; i += 1024) {
        int r  = i >> 10;
        int kk = i & 1023;
        v4f x4 = *(const v4f*)(X + (size_t)(mtile * 16 + r) * NFEAT + kk);
        union { __bf16 h[4]; unsigned long long u; } pk;
        #pragma unroll
        for (int e = 0; e < 4; ++e) pk.h[e] = (__bf16)x4[e];
        *(unsigned long long*)(As + i) = pk.u;
    }
    __syncthreads();

    const int wave = t >> 5;
    const int lane = t & 31;
    const int hi   = lane >> 4;          // 0: lanes 0-15, 1: lanes 16-31
    const int l    = lane & 15;

    const int n0 = wave * 16 + l;            // first  N-tile column
    const int n1 = (wave + 8) * 16 + l;      // second N-tile column

    // A frag (16x32 bf16): lanes 0-15 K:0-15, lanes 16-31 K:16-31
    const __bf16* ar = As + l * NFEAT + (hi ? 16 : 0);
    // B frag (32x16 bf16): lanes 0-15 K:0-15, lanes 16-31 K:16-31 of column n
    const __bf16* w0 = Wt + (size_t)n0 * NFEAT + (hi ? 16 : 0);
    const __bf16* w1 = Wt + (size_t)n1 * NFEAT + (hi ? 16 : 0);

    v8f acc0 = {};
    v8f acc1 = {};
    #pragma unroll 4
    for (int k0 = 0; k0 < NFEAT; k0 += 32) {
        __builtin_prefetch(w0 + k0 + 128, 0, 1);
        __builtin_prefetch(w1 + k0 + 128, 0, 1);
        v8bf alo = *(const v8bf*)(ar + k0);
        v8bf ahi = *(const v8bf*)(ar + k0 + 8);
        v16bf a;
        #pragma unroll
        for (int e = 0; e < 8; ++e) { a[e] = alo[e]; a[8 + e] = ahi[e]; }
        v16bf b0 = *(const v16bf*)(w0 + k0);
        v16bf b1 = *(const v16bf*)(w1 + k0);
        acc0 = __builtin_amdgcn_wmma_f32_16x16x32_bf16(
                   false, a, false, b0, (short)0, acc0, false, false);
        acc1 = __builtin_amdgcn_wmma_f32_16x16x32_bf16(
                   false, a, false, b1, (short)0, acc1, false, false);
    }

    // C/D layout: lanes 0-15: (M=r, N=lane) ; lanes 16-31: (M=8+r, N=lane-16)
    const float bn0 = bias[n0];
    const float bn1 = bias[n1];
    #pragma unroll
    for (int r = 0; r < 8; ++r) {
        int row = mtile * 16 + r + (hi ? 8 : 0);
        H[(size_t)row * DDIM + n0] = fmaxf(acc0[r] + bn0, 0.f);
        H[(size_t)row * DDIM + n1] = fmaxf(acc1[r] + bn1, 0.f);
    }
}

// ---------------------------------------------------------------------------
// Kernel 2: per-layer prep -> xn = l2norm(per-capsule h); layer0 also applies
// per-capsule FC + relu + renorm.  1 block / node, thread t -> (k=t/32,d=t%32)
// ---------------------------------------------------------------------------
__global__ void k_prep(const float* __restrict__ h,
                       float* __restrict__ xn,
                       const float* __restrict__ fcw,   // [K][DDo][DDi]
                       const float* __restrict__ fcb,   // [K][DD]
                       int use_fc) {
    __shared__ float xl[DDIM];
    const int node = blockIdx.x;
    const int t = threadIdx.x;
    const int k = t >> 5;
    const int d = t & 31;

    float v = h[(size_t)node * DDIM + t];
    float ss = wave_sum(v * v);
    float xd = v / fmaxf(sqrtf(ss), 1e-12f);

    float out;
    if (use_fc) {
        xl[t] = xd;
        __syncthreads();
        float y = fcb[k * DD + d];
        const float* wrow = fcw + ((size_t)(k * DD + d)) * DD;  // fc_w[k][o=d][:]
        #pragma unroll 8
        for (int i = 0; i < DD; ++i) y += xl[k * DD + i] * wrow[i];
        y = fmaxf(y, 0.f);
        float ss2 = wave_sum(y * y);
        out = y / fmaxf(sqrtf(ss2), 1e-12f);
    } else {
        out = xd;
    }
    xn[(size_t)node * DDIM + t] = out;
}

// ---------------------------------------------------------------------------
// Kernel 3: full routing for one node (all iterations in-kernel, z in LDS).
//   Neighbor gather done by the Tensor Data Mover in gather mode:
//   16 x (256 x f32) rows selected by 16-bit row indices -> packed into LDS.
// ---------------------------------------------------------------------------
__global__ void k_routing(const float* __restrict__ xn,
                          float* __restrict__ hout,
                          const int* __restrict__ nb,
                          const float* __restrict__ raw_param,
                          const int* __restrict__ routit,
                          int do_relu) {
    __shared__ float zs[MM * DDIM];      // 16 KB: z[m][k][d] at m*256 + t
    __shared__ float pS[MM * KC];
    __shared__ float p1S[MM * KC];
    __shared__ float p2S[MM * KC];

    const int node = blockIdx.x;
    const int t = threadIdx.x;
    const int k = t >> 5;
    const int d = t & 31;

    const float par = 1.f / (1.f + __expf(-raw_param[0]));
    const int T = *routit;

#if HAVE_TDM
    // ---- TDM gather: one wave builds D# and issues tensor_load_to_lds ----
    if (t < 32) {
        unsigned idx[MM];
        #pragma unroll
        for (int i = 0; i < MM; ++i)
            idx[i] = (unsigned)nb[(size_t)node * MM + i] & 0xFFFFu;

        unsigned long long gaddr = (unsigned long long)(size_t)xn;
        unsigned lds_off = (unsigned)(size_t)(void*)zs;

        v4u g0;
        g0[0] = 0x80000001u;                       // count=1, gather_mode=1, 16-bit idx
        g0[1] = lds_off;                           // LDS destination
        g0[2] = (unsigned)gaddr;                   // global addr [31:0]
        g0[3] = ((unsigned)(gaddr >> 32) & 0x01FFFFFFu) | (2u << 30); // [56:32] | type=2

        v8i g1;
        g1[0] = (int)(2u << 16);                   // data_size = 4 bytes
        g1[1] = (int)((DDIM & 0xFFFFu) << 16);     // tensor_dim0[15:0] = 256
        g1[2] = (int)(((unsigned)NN & 0xFFFFu) << 16); // tensor_dim1[15:0] = 30000
        g1[3] = (int)(((unsigned)DDIM) << 16);     // tile_dim0 = 256 (row width)
        g1[4] = MM;                                // tile_dim1 = 16 valid indices
        g1[5] = DDIM;                              // tensor_dim0_stride = 256 elems
        g1[6] = 0;
        g1[7] = 0;

        v4i g2, g3;
        #pragma unroll
        for (int w = 0; w < 4; ++w) {
            g2[w] = (int)((idx[2 * w + 1] << 16) | idx[2 * w]);
            g3[w] = (int)((idx[8 + 2 * w + 1] << 16) | idx[8 + 2 * w]);
        }

#if defined(__clang_major__) && (__clang_major__ >= 23)
        v8i gz = {};
        __builtin_amdgcn_tensor_load_to_lds(g0, g1, g2, g3, gz, 0);
#else
        __builtin_amdgcn_tensor_load_to_lds(g0, g1, g2, g3, 0);
#endif
        __builtin_amdgcn_s_wait_tensorcnt(0);
    }
#else
    // ---- fallback: per-thread gather ----
    for (int mi = 0; mi < MM; ++mi) {
        int j = nb[(size_t)node * MM + mi];
        float v = 0.f;
        if ((unsigned)j < (unsigned)NN) v = xn[(size_t)j * DDIM + t];
        zs[mi * DDIM + t] = v;
    }
#endif

    float ub = xn[(size_t)node * DDIM + t];   // u_before (normalized x)
    float un = ub;                            // normalized u (used for p)
    __syncthreads();

    for (int it = 0; it < T; ++it) {
        if (it > 0) {
            // p[m][k] = sum_d z[m][k][d] * u[k][d]   (wave k reduces over d)
            #pragma unroll
            for (int m = 0; m < MM; ++m) {
                float v = zs[m * DDIM + t] * un;
                v = wave_sum(v);
                if (d == 0) pS[m * KC + k] = v;
            }
        } else {
            if (t < MM * KC) pS[t] = 0.f;     // first iter: p = 0
        }
        __syncthreads();

        // softmax over neighbors (axis m), per capsule k : threads 0..7
        if (t < KC) {
            int kk = t;
            float mx = -1e30f;
            for (int m = 0; m < MM; ++m) mx = fmaxf(mx, pS[m * KC + kk]);
            float s = 0.f;
            for (int m = 0; m < MM; ++m) s += __expf(pS[m * KC + kk] - mx);
            float inv = 1.f / s;
            for (int m = 0; m < MM; ++m)
                p1S[m * KC + kk] = __expf(pS[m * KC + kk] - mx) * inv;
        } else if (t < KC + MM) {
            // softmax over capsules (axis k), per neighbor m : threads 8..23
            int m = t - KC;
            float mx = -1e30f;
            for (int kk = 0; kk < KC; ++kk) mx = fmaxf(mx, pS[m * KC + kk]);
            float s = 0.f;
            for (int kk = 0; kk < KC; ++kk) s += __expf(pS[m * KC + kk] - mx);
            float inv = 1.f / s;
            for (int kk = 0; kk < KC; ++kk)
                p2S[m * KC + kk] = __expf(pS[m * KC + kk] - mx) * inv;
        }
        __syncthreads();

        // u1/u2 accumulation + update
        float a1 = 0.f, a2 = 0.f;
        #pragma unroll
        for (int m = 0; m < MM; ++m) {
            float zz = zs[m * DDIM + t];
            a1 += zz * p1S[m * KC + k];
            a2 += zz * p2S[m * KC + k];
        }
        float unew = par * a1 + (1.f - par) * a2 + ub;
        ub = unew;
        if (it < T - 1) {
            float ss = wave_sum(unew * unew);
            un = unew / fmaxf(sqrtf(ss), 1e-12f);
        }
        __syncthreads();
    }

    float out = ub;                            // final u is NOT normalized
    if (do_relu) out = fmaxf(out, 0.f);
    hout[(size_t)node * DDIM + t] = out;
}

// ---------------------------------------------------------------------------
// Kernel 4: logits = h @ mlp_w^T + b ; log_softmax ; emit (log_probs, h)
// ---------------------------------------------------------------------------
__global__ void k_final(const float* __restrict__ h,
                        const float* __restrict__ mw,
                        const float* __restrict__ mb,
                        float* __restrict__ out) {
    __shared__ float hl[DDIM];
    __shared__ float lg[NCLASS];
    __shared__ float red[2];
    const int row = blockIdx.x;
    const int t = threadIdx.x;

    float v = h[(size_t)row * DDIM + t];
    hl[t] = v;
    out[(size_t)NN * NCLASS + (size_t)row * DDIM + t] = v;   // embedding output
    __syncthreads();

    if (t < NCLASS) {
        float a = mb[t];
        const float* wr = mw + (size_t)t * DDIM;
        #pragma unroll 8
        for (int j = 0; j < DDIM; ++j) a += hl[j] * wr[j];
        lg[t] = a;
    }
    __syncthreads();
    if (t == 0) {
        float mx = -1e30f;
        for (int c = 0; c < NCLASS; ++c) mx = fmaxf(mx, lg[c]);
        float s = 0.f;
        for (int c = 0; c < NCLASS; ++c) s += __expf(lg[c] - mx);
        red[0] = mx;
        red[1] = __logf(s);
    }
    __syncthreads();
    if (t < NCLASS)
        out[(size_t)row * NCLASS + t] = lg[t] - red[0] - red[1];
}

// ---------------------------------------------------------------------------
extern "C" void kernel_launch(void* const* d_in, const int* in_sizes, int n_in,
                              void* d_out, int out_size, void* d_ws, size_t ws_size,
                              hipStream_t stream) {
    const float* x         = (const float*)d_in[0];
    const int*   neighbors = (const int*)  d_in[1];
    const float* pca_w     = (const float*)d_in[2];
    const float* pca_b     = (const float*)d_in[3];
    const float* raw_param = (const float*)d_in[4];
    const float* fc_w      = (const float*)d_in[5];
    const float* fc_b      = (const float*)d_in[6];
    const float* mlp_w     = (const float*)d_in[7];
    const float* mlp_b     = (const float*)d_in[8];
    const int*   routit    = (const int*)  d_in[9];
    float* out = (float*)d_out;

    char* ws = (char*)d_ws;
    const size_t WT_BYTES = (size_t)DDIM * NFEAT * sizeof(__bf16);   // 512 KB
    const size_t H_BYTES  = (size_t)NN * DDIM * sizeof(float);       // 30.72 MB
    __bf16* Wt = (__bf16*)ws;
    float*  h  = (float*)(ws + WT_BYTES);
    float*  xn = (float*)(ws + WT_BYTES + H_BYTES);

    k_conv_wt <<<(DDIM * NFEAT) / 256, 256, 0, stream>>>(pca_w, Wt);
    k_pca_gemm<<<NN / 16,              256, 0, stream>>>(x, Wt, pca_b, h);

    for (int layer = 0; layer < NLAYER; ++layer) {
        k_prep   <<<NN, 256, 0, stream>>>(h, xn, fc_w, fc_b, layer == 0 ? 1 : 0);
        k_routing<<<NN, 256, 0, stream>>>(xn, h, neighbors, raw_param, routit,
                                          layer < NLAYER - 1 ? 1 : 0);
    }

    k_final<<<NN, 256, 0, stream>>>(h, mlp_w, mlp_b, out);
}
